// EmbeddingAlignmentGNN_45122926412247
// MI455X (gfx1250) — compile-verified, matrix operand order checked by reading, not run on hardware
//
#include <hip/hip_runtime.h>
#include <hip/hip_bf16.h>

typedef _Float16 v16h __attribute__((ext_vector_type(16)));
typedef float    v8f  __attribute__((ext_vector_type(8)));

#define IN_DIM  256
#define OUT_DIM 128

// ---------------------------------------------------------------------------
// 1) Fuse the two linear layers: Wc[n][k] = sum_m Wg[n][m] * Wp[m][k], f16 out
//    (h = (x @ Wp^T) @ Wg^T == x @ (Wg @ Wp)^T)
// ---------------------------------------------------------------------------
__global__ void k_fuse_w(const float* __restrict__ Wp,   // [OUT][IN]
                         const float* __restrict__ Wg,   // [OUT][OUT]
                         _Float16* __restrict__ Wc)      // [OUT][IN] f16
{
    const int n = blockIdx.x;     // 0..OUT_DIM-1
    const int k = threadIdx.x;    // 0..IN_DIM-1
    float acc = 0.0f;
#pragma unroll 8
    for (int m = 0; m < OUT_DIM; ++m)
        acc += Wg[n * OUT_DIM + m] * Wp[m * IN_DIM + k];
    Wc[n * IN_DIM + k] = (_Float16)acc;
}

// ---------------------------------------------------------------------------
// 2) Degree: init to 1 (self loop), count targets, then dinv = rsqrt(deg)
// ---------------------------------------------------------------------------
__global__ void k_deg_init(float* __restrict__ deg, int n)
{
    int i = blockIdx.x * blockDim.x + threadIdx.x;
    if (i < n) deg[i] = 1.0f;
}

__global__ void k_deg_count(const int* __restrict__ col, float* __restrict__ deg, int e)
{
    int i = blockIdx.x * blockDim.x + threadIdx.x;
    if (i < e) atomicAdd(&deg[col[i]], 1.0f);
}

__global__ void k_dinv(float* __restrict__ deg, int n)
{
    int i = blockIdx.x * blockDim.x + threadIdx.x;
    if (i < n) deg[i] = rsqrtf(deg[i]);   // deg >= 1 always (self loop)
}

// ---------------------------------------------------------------------------
// 3) WMMA GEMM: h[N_nodes x 128] = x[N_nodes x 256] (f32->f16) @ Wc^T (f16)
//    Block = 256 threads = 8 waves; block owns 16 rows, wave w owns N-tile w.
//    A fragment (16-bit 16x32): lane L -> row M=L%16; khalf=L/16 selects
//    K groups {0..7,16..23} vs {8..15,24..31}; pairs packed per VGPR.
//    B fragment: same K packing with lane -> N column (Wc stored [N][K]).
// ---------------------------------------------------------------------------
__global__ __launch_bounds__(256) void
k_gemm(const float* __restrict__ x,      // [N][256] f32
       const _Float16* __restrict__ Wc,  // [128][256] f16
       float* __restrict__ h,            // [N][128] f32
       int nNodes)
{
    const int lid   = threadIdx.x & 31;
    const int wave  = threadIdx.x >> 5;   // 0..7 -> N tile
    const int lidlo = lid & 15;
    const int khalf = lid >> 4;           // 0 or 1

    // A: this lane loads row (clamped so EXEC stays all-ones for WMMA)
    int arow = blockIdx.x * 16 + lidlo;
    int arc  = arow < nNodes ? arow : nNodes - 1;
    const float*    xrow = x  + (size_t)arc * IN_DIM;
    // B: this lane loads column wave*16+lidlo of Wc ([N][K] row-major)
    const _Float16* wrow = Wc + (size_t)(wave * 16 + lidlo) * IN_DIM;

    v8f c = {};
#pragma unroll
    for (int kb = 0; kb < IN_DIM; kb += 32) {
        union { v16h v; _Float16 f[16]; unsigned u[8]; } a, b;
#pragma unroll
        for (int r = 0; r < 8; ++r) {
            int kk = kb + ((r < 4) ? (khalf * 8 + 2 * r)
                                   : (16 + khalf * 8 + 2 * (r - 4)));
            float2 a2 = *(const float2*)(xrow + kk);
            a.f[2 * r]     = (_Float16)a2.x;
            a.f[2 * r + 1] = (_Float16)a2.y;
            b.u[r] = *(const unsigned*)(wrow + kk);
        }
        c = __builtin_amdgcn_wmma_f32_16x16x32_f16(
                /*neg_a=*/false, a.v, /*neg_b=*/false, b.v,
                /*c_mod=*/(short)0, c, /*reuse_a=*/false, /*reuse_b=*/false);
    }

    // D layout: VGPR r -> M = r + 8*khalf; N = lidlo
#pragma unroll
    for (int r = 0; r < 8; ++r) {
        int m = blockIdx.x * 16 + r + 8 * khalf;
        if (m < nNodes)
            h[(size_t)m * OUT_DIM + wave * 16 + lidlo] = c[r];
    }
}

// ---------------------------------------------------------------------------
// 4) Self-loop init: out[i][j] = h[i][j] * dinv[i]^2
// ---------------------------------------------------------------------------
__global__ void k_self(const float* __restrict__ h, const float* __restrict__ dinv,
                       float* __restrict__ out, int total)
{
    int i = blockIdx.x * blockDim.x + threadIdx.x;
    if (i < total) {
        int node = i >> 7;                 // /OUT_DIM
        float d = dinv[node];
        out[i] = h[i] * d * d;
    }
}

// ---------------------------------------------------------------------------
// 5) Edge scatter: one wave32 per edge, 4 floats per lane, f32 atomics
// ---------------------------------------------------------------------------
__global__ __launch_bounds__(256) void
k_scatter(const float* __restrict__ h, const int* __restrict__ ei,
          const float* __restrict__ dinv, float* __restrict__ out, int nEdges)
{
    int gid  = blockIdx.x * blockDim.x + threadIdx.x;
    int edge = gid >> 5;
    int lane = gid & 31;
    if (edge >= nEdges) return;

    int src = ei[edge];
    int dst = ei[nEdges + edge];
    float w = dinv[src] * dinv[dst];

    float4 m = *(const float4*)(h + (size_t)src * OUT_DIM + lane * 4);
    float* od = out + (size_t)dst * OUT_DIM + lane * 4;
    atomicAdd(od + 0, m.x * w);
    atomicAdd(od + 1, m.y * w);
    atomicAdd(od + 2, m.z * w);
    atomicAdd(od + 3, m.w * w);
}

// ---------------------------------------------------------------------------
// 6) Bias + row L2 normalize (one wave per node, shuffle reduction)
// ---------------------------------------------------------------------------
__global__ __launch_bounds__(256) void
k_norm(float* __restrict__ out, const float* __restrict__ bg, int nNodes)
{
    int node = blockIdx.x * 8 + (threadIdx.x >> 5);
    int lane = threadIdx.x & 31;
    if (node >= nNodes) return;

    float4 v = *(const float4*)(out + (size_t)node * OUT_DIM + lane * 4);
    float4 b = *(const float4*)(bg + lane * 4);
    v.x += b.x; v.y += b.y; v.z += b.z; v.w += b.w;

    float ss = v.x * v.x + v.y * v.y + v.z * v.z + v.w * v.w;
#pragma unroll
    for (int off = 16; off >= 1; off >>= 1)
        ss += __shfl_xor(ss, off, 32);

    float inv = 1.0f / fmaxf(sqrtf(ss), 1e-12f);
    v.x *= inv; v.y *= inv; v.z *= inv; v.w *= inv;
    *(float4*)(out + (size_t)node * OUT_DIM + lane * 4) = v;
}

// ---------------------------------------------------------------------------
// Launch
// ---------------------------------------------------------------------------
extern "C" void kernel_launch(void* const* d_in, const int* in_sizes, int n_in,
                              void* d_out, int out_size, void* d_ws, size_t ws_size,
                              hipStream_t stream)
{
    const float* x  = (const float*)d_in[0];   // [N][256]
    const int*   ei = (const int*)d_in[1];     // [2][E]
    const float* Wp = (const float*)d_in[2];   // [128][256]
    const float* Wg = (const float*)d_in[3];   // [128][128]
    const float* bg = (const float*)d_in[4];   // [128]
    float* out = (float*)d_out;

    const int nNodes = in_sizes[0] / IN_DIM;
    const int nEdges = in_sizes[1] / 2;

    // workspace carve-out
    char* ws = (char*)d_ws;
    _Float16* Wc = (_Float16*)ws;
    size_t off = ((size_t)OUT_DIM * IN_DIM * sizeof(_Float16) + 255) & ~(size_t)255;
    float* h = (float*)(ws + off);
    off += (((size_t)nNodes * OUT_DIM * sizeof(float)) + 255) & ~(size_t)255;
    float* deg = (float*)(ws + off);   // deg then dinv (in place)

    k_fuse_w<<<OUT_DIM, IN_DIM, 0, stream>>>(Wp, Wg, Wc);

    k_deg_init<<<(nNodes + 255) / 256, 256, 0, stream>>>(deg, nNodes);
    k_deg_count<<<(nEdges + 255) / 256, 256, 0, stream>>>(ei + nEdges, deg, nEdges);
    k_dinv<<<(nNodes + 255) / 256, 256, 0, stream>>>(deg, nNodes);

    k_gemm<<<(nNodes + 15) / 16, 256, 0, stream>>>(x, Wc, h, nNodes);

    const int total = nNodes * OUT_DIM;
    k_self<<<(total + 255) / 256, 256, 0, stream>>>(h, deg, out, total);

    const int edgesPerBlock = 256 / 32;
    k_scatter<<<(nEdges + edgesPerBlock - 1) / edgesPerBlock, 256, 0, stream>>>(
        h, ei, deg, out, nEdges);

    k_norm<<<(nNodes + 7) / 8, 256, 0, stream>>>(out, bg, nNodes);
}
